// BM_layer_onehalf_90812788507083
// MI455X (gfx1250) — compile-verified, matrix operand order checked by reading, not run on hardware
//
#include <hip/hip_runtime.h>
#include <hip/hip_bf16.h>
#include <cstdint>

typedef __attribute__((ext_vector_type(16))) _Float16 v16h;
typedef __attribute__((ext_vector_type(8)))  _Float16 v8h;
typedef __attribute__((ext_vector_type(8)))  float    v8f;

#define BATCH 8
#define CIN   32
#define HH    28
#define WW    28
#define HP    30                  // padded
#define WP    30
#define COUT  64
#define MTOT  (BATCH * HH * WW)   // 6272
#define KTOT  288                 // 9 * 32
#define MTILE 64
#define BLOCK 128                 // 4 waves
#define WPACK_HALVES (9 * 4 * 32 * 16)        // 18432 halves = 36864 B
#define UPACK_HALVES (BATCH * HP * WP * CIN)  // 230400 halves = 460800 B

// ---------------- prep 1: pack W' = exp(kf+5) - delta_w into B-fragment order --
// layout index = ((tap*4 + t)*32 + lane)*16 + e
// lane -> (n = t*16 + (lane&15), khalf = lane>>4)
// e    -> kk = (e<8) ? 8*khalf+e : 16 + 8*khalf + (e-8)   (ISA 16-bit layout)
__global__ void pack_w_kernel(const float* __restrict__ kf,
                              const float* __restrict__ dw,
                              _Float16* __restrict__ wpack) {
    int idx  = blockIdx.x * 256 + threadIdx.x;   // 0 .. 18431
    int e    = idx & 15;
    int lane = (idx >> 4) & 31;
    int t    = (idx >> 9) & 3;
    int tap  = idx >> 11;                        // 0..8
    int n    = t * 16 + (lane & 15);
    int kh   = lane >> 4;
    int kk   = (e < 8) ? (8 * kh + e) : (16 + 8 * kh + (e - 8));
    int r    = tap * 32 + kk;                    // row in kf (288 x 64, row-major)
    float w  = __expf(kf[r * COUT + n] + 5.0f) - dw[0];
    wpack[idx] = (_Float16)w;
}

// ---------------- prep 2: u = x+1, padded border = 1, channel-last f16 --------
// up[b][ih][iw][ci]  (8,30,30,32)
__global__ void pack_u_kernel(const float* __restrict__ x,
                              _Float16* __restrict__ up) {
    int idx = blockIdx.x * 256 + threadIdx.x;    // 0 .. 230399
    int ci  = idx & 31;
    int p   = idx >> 5;                          // (b*30+ih)*30+iw
    int iw  = p % WP;
    int t   = p / WP;
    int ih  = t % HP;
    int b   = t / HP;
    int xh = ih - 1, xw = iw - 1;
    float v = 1.0f;
    if (xh >= 0 && xh < HH && xw >= 0 && xw < WW)
        v = x[(((size_t)b * CIN + ci) * HH + xh) * WW + xw] + 1.0f;
    up[idx] = (_Float16)v;
}

// per-channel epilogue constant: C[c] = 288*dw - dx * sum_r kf[r,c] + bias[c]
__global__ void make_c_kernel(const float* __restrict__ kf,
                              const float* __restrict__ bias,
                              const float* __restrict__ dx,
                              const float* __restrict__ dw,
                              float* __restrict__ cvec) {
    int n = threadIdx.x;                         // 64 threads
    float s = 0.f;
    for (int r = 0; r < KTOT; ++r) s += kf[r * COUT + n];
    cvec[n] = 288.0f * dw[0] - dx[0] * s + bias[n];
}

// ---------------- main fused im2col GEMM (barrier-free inner loop) ------------
__global__ __launch_bounds__(BLOCK) void fused_gemm_kernel(
    const _Float16* __restrict__ up,       // (8,30,30,32) f16, u=x+1 padded
    const _Float16* __restrict__ wpack,    // packed B fragments (36864 B)
    const float*    __restrict__ cvec,     // (64)
    float*          __restrict__ out) {    // (8,64,28,28)

    __shared__ __align__(32) _Float16 wsh[WPACK_HALVES];  // 36 KB
    __shared__ float csh[COUT];

    const int tid  = threadIdx.x;
    const int lane = tid & 31;
    const int wv   = tid >> 5;               // 0..3
    const int m0   = blockIdx.x * MTILE;

    // ---- async DMA of packed weights to LDS (ASYNCcnt path) ----
    {
        unsigned lds0 = (unsigned)(uintptr_t)(&wsh[0]);   // low 32 bits = LDS offset
        unsigned long long g0 = (unsigned long long)(const void*)wpack;
        #pragma unroll
        for (int j = 0; j < 18; ++j) {                    // 2304 x 16B / 128 thr
            unsigned ofs = (unsigned)(tid + j * BLOCK) * 16u;
            unsigned laddr = lds0 + ofs;
            unsigned long long ga = g0 + (unsigned long long)ofs;
            asm volatile("global_load_async_to_lds_b128 %0, %1, off"
                         :: "v"(laddr), "v"(ga) : "memory");
        }
    }
    if (tid < COUT) csh[tid] = cvec[tid];
    asm volatile("s_wait_asynccnt 0" ::: "memory");
    __syncthreads();

    // A-fragment base: lane -> (m = row, khalf selects K {0..7,16..23} vs {8..15,24..31})
    const int m     = m0 + wv * 16 + (lane & 15);
    const int khalf = lane >> 4;
    const int bb    = m / (HH * WW);
    const int hw    = m % (HH * WW);
    const int oh    = hw / WW;
    const int ow    = hw % WW;
    // position in padded u: tap (kh,kw) reads (oh+kh, ow+kw)
    const _Float16* abase =
        up + ((((size_t)bb * HP + oh) * WP + ow) << 5) + 8 * khalf;

    v8f acc[4] = {v8f{}, v8f{}, v8f{}, v8f{}};

    #pragma unroll
    for (int tap = 0; tap < 9; ++tap) {
        const int kh = tap / 3, kw = tap % 3;
        const _Float16* ap = abase + (kh * WP + kw) * CIN;  // static imm offsets
        v8h alo = *(const v8h*)(ap);        // channels 8*khalf   .. +8
        v8h ahi = *(const v8h*)(ap + 16);   // channels 16+8*khalf.. +8
        v16h a  = __builtin_shufflevector(alo, ahi,
                    0,1,2,3,4,5,6,7,8,9,10,11,12,13,14,15);

        const _Float16* wp = &wsh[tap * 2048 + lane * 16];
        #pragma unroll
        for (int t = 0; t < 4; ++t) {
            v16h b = *(const v16h*)(wp + t * 512);
            acc[t] = __builtin_amdgcn_wmma_f32_16x16x32_f16(
                         false, a, false, b, (short)0, acc[t], false, false);
        }
    }

    // ---- epilogue: add per-channel constant, store 32B-aligned v8f runs ----
    // C/D layout: vgpr v -> M = 8*(lane>>4) + v ; N = t*16 + (lane&15)
    const int mb  = m0 + wv * 16 + ((lane >> 4) * 8);     // multiple of 8
    const int b2  = mb / (HH * WW);
    const int hw2 = mb % (HH * WW);                       // multiple of 8 -> 32B aligned
    #pragma unroll
    for (int t = 0; t < 4; ++t) {
        const int c = t * 16 + (lane & 15);
        v8f r = acc[t] + csh[c];
        *(v8f*)&out[((size_t)b2 * COUT + c) * (HH * WW) + hw2] = r;
    }
}

extern "C" void kernel_launch(void* const* d_in, const int* in_sizes, int n_in,
                              void* d_out, int out_size, void* d_ws, size_t ws_size,
                              hipStream_t stream) {
    const float* x    = (const float*)d_in[0];   // (8,32,28,28)
    const float* kf   = (const float*)d_in[1];   // (3,3,32,64) == (288,64) row-major
    const float* bias = (const float*)d_in[2];   // (64,)
    const float* dx   = (const float*)d_in[3];   // scalar
    const float* dw   = (const float*)d_in[4];   // scalar

    _Float16* wpack = (_Float16*)d_ws;                               // 36864 B
    float*    cvec  = (float*)((char*)d_ws + WPACK_HALVES * 2);      // +256 B
    _Float16* up    = (_Float16*)((char*)d_ws + WPACK_HALVES * 2 + 512);

    pack_w_kernel<<<WPACK_HALVES / 256, 256, 0, stream>>>(kf, dw, wpack);
    pack_u_kernel<<<UPACK_HALVES / 256, 256, 0, stream>>>(x, up);
    make_c_kernel<<<1, COUT, 0, stream>>>(kf, bias, dx, dw, cvec);
    fused_gemm_kernel<<<MTOT / MTILE, BLOCK, 0, stream>>>(
        up, wpack, cvec, (float*)d_out);
}
